// DiscRNNG_23252952941013
// MI455X (gfx1250) — compile-verified, matrix operand order checked by reading, not run on hardware
//
#include <hip/hip_runtime.h>
#include <hip/hip_bf16.h>
#include <vector>

// ---------------------------------------------------------------------------
// DiscRNNG forward on MI455X (gfx1250, wave32).
// All GEMMs (LSTM cells, MLPs, projections) run on v_wmma_f32_16x16x32_bf16.
// Host code fully unrolls the (deterministic) shift-reduce action sequence.
// ---------------------------------------------------------------------------

typedef __bf16 bf16_t;
typedef __attribute__((ext_vector_type(16))) __bf16 v16bf;
typedef __attribute__((ext_vector_type(8)))  float  v8f;

#define BSZ   512
#define SLEN  40
#define WDIM  256
#define ADIM  64
#define HID   128
#define NACT  66
#define NACTP 80          // padded to WMMA N tiles
#define ALEN  52
#define BH    (BSZ*HID)   // 65536 elements

// ---------------------------------------------------------------------------
// Device helpers
// ---------------------------------------------------------------------------
__device__ __forceinline__ float sigm_(float x) { return 1.0f / (1.0f + __expf(-x)); }

// Load one 16x32 bf16 fragment (rows x K-chunk), row-major with leading dim ld.
// Lane layout per CDNA5 ISA 7.12.2 (16-bit A/B fragment): row = lane%16,
// upper half-wave takes K+8; VGPR v holds K pair ((v&3)*2 + (v>>2)*16).
__device__ __forceinline__ v16bf load_frag_k32(const bf16_t* base, int ld) {
  int lane = threadIdx.x & 31;
  int r    = lane & 15;
  int kb   = (lane & 16) ? 8 : 0;
  union { v16bf v; unsigned u[8]; } f;
  const bf16_t* p = base + (size_t)r * ld + kb;
#pragma unroll
  for (int i = 0; i < 8; ++i) {
    int k = ((i & 3) * 2) + ((i >> 2) * 16);
    f.u[i] = *(const unsigned*)(p + k);   // 2 bf16, 4B aligned
  }
  return f.v;
}

// ---------------------------------------------------------------------------
// Fused LSTM step: gates = x @ Wih^T + h @ Whh^T + bih + bhh (4H = 512 cols),
// then PyTorch-order i,f,g,o nonlinearity + c/h update.
// Grid (8, 8), block 128 (4 waves). Wave owns 16 batch rows x 16 h-cols,
// with 4 accumulators (one per gate quadrant).
// ---------------------------------------------------------------------------
__global__ void lstm_step_wmma(const bf16_t* __restrict__ x, int Kx,
                               const bf16_t* __restrict__ hprev,
                               const bf16_t* __restrict__ Wih,
                               const bf16_t* __restrict__ Whh,
                               const float*  __restrict__ bih,
                               const float*  __restrict__ bhh,
                               float*        __restrict__ c,
                               bf16_t*       __restrict__ hout) {
  const int mbase = blockIdx.x * 64 + (threadIdx.x >> 5) * 16;
  const int hc    = blockIdx.y * 16;
  v8f z = {};
  v8f acc[4] = { z, z, z, z };

  for (int k = 0; k < Kx; k += 32) {
    v16bf a = load_frag_k32(x + (size_t)mbase * Kx + k, Kx);
#pragma unroll
    for (int g = 0; g < 4; ++g) {
      v16bf w = load_frag_k32(Wih + (size_t)(g * HID + hc) * Kx + k, Kx);
      acc[g] = __builtin_amdgcn_wmma_f32_16x16x32_bf16(false, a, false, w,
                                                       (short)0, acc[g], false, false);
    }
  }
  for (int k = 0; k < HID; k += 32) {
    v16bf a = load_frag_k32(hprev + (size_t)mbase * HID + k, HID);
#pragma unroll
    for (int g = 0; g < 4; ++g) {
      v16bf w = load_frag_k32(Whh + (size_t)(g * HID + hc) * HID + k, HID);
      acc[g] = __builtin_amdgcn_wmma_f32_16x16x32_bf16(false, a, false, w,
                                                       (short)0, acc[g], false, false);
    }
  }

  const int lane   = threadIdx.x & 31;
  const int n      = hc + (lane & 15);
  const int rowoff = (lane & 16) ? 8 : 0;
  float bb[4];
#pragma unroll
  for (int g = 0; g < 4; ++g) bb[g] = bih[g * HID + n] + bhh[g * HID + n];

#pragma unroll
  for (int r = 0; r < 8; ++r) {
    int m = mbase + r + rowoff;
    float gi = sigm_(acc[0][r] + bb[0]);
    float gf = sigm_(acc[1][r] + bb[1]);
    float gg = tanhf(acc[2][r] + bb[2]);
    float go = sigm_(acc[3][r] + bb[3]);
    size_t idx = (size_t)m * HID + n;
    float cn = gf * c[idx] + gi * gg;
    c[idx]   = cn;
    hout[idx] = (bf16_t)(go * tanhf(cn));
  }
}

// ---------------------------------------------------------------------------
// Generic GEMM: out[M=512, n] = act(A[512,K] @ W[N,K]^T + bias[n]).
// Grid (8, ceil(N/16)), block 128. OUT in {bf16, f32}.
// ---------------------------------------------------------------------------
template <typename OUT, bool RELU>
__global__ void gemm_wmma(const bf16_t* __restrict__ A,
                          const bf16_t* __restrict__ W,
                          const float*  __restrict__ bias,
                          OUT*          __restrict__ out,
                          int K, int Nvalid, int ldc) {
  const int mbase = blockIdx.x * 64 + (threadIdx.x >> 5) * 16;
  const int nbase = blockIdx.y * 16;
  v8f acc = {};
  for (int k = 0; k < K; k += 32) {
    v16bf a = load_frag_k32(A + (size_t)mbase * K + k, K);
    v16bf b = load_frag_k32(W + (size_t)nbase * K + k, K);
    acc = __builtin_amdgcn_wmma_f32_16x16x32_bf16(false, a, false, b,
                                                  (short)0, acc, false, false);
  }
  const int lane   = threadIdx.x & 31;
  const int n      = nbase + (lane & 15);
  const int rowoff = (lane & 16) ? 8 : 0;
  if (n >= Nvalid) return;
  float bv = bias ? bias[n] : 0.0f;
#pragma unroll
  for (int r = 0; r < 8; ++r) {
    int m = mbase + r + rowoff;
    float v = acc[r] + bv;
    if (RELU) v = v > 0.0f ? v : 0.0f;
    out[(size_t)m * ldc + n] = (OUT)v;
  }
}

// ---------------------------------------------------------------------------
// Small utility kernels
// ---------------------------------------------------------------------------
__global__ void k_f32_to_bf16(bf16_t* dst, const float* src, int n) {
  int i = blockIdx.x * 256 + threadIdx.x;
  if (i < n) dst[i] = (bf16_t)src[i];
}

// convert [rows_src, cols] f32 -> [rows_dst, cols] bf16 with zero pad rows
__global__ void k_conv_pad(bf16_t* dst, const float* src, int rows_src, int rows_dst, int cols) {
  int i = blockIdx.x * 256 + threadIdx.x;
  int n = rows_dst * cols;
  if (i < n) {
    int r = i / cols;
    dst[i] = (r < rows_src) ? (bf16_t)src[i] : (bf16_t)0.0f;
  }
}

// buff[t,b,:] = word_emb[words[b, SLEN-1-t], :]  (reversed buffer)
__global__ void k_gather_words(bf16_t* dst, const int* words, const float* emb) {
  int t = blockIdx.x, b = blockIdx.y, cix = threadIdx.x;
  int w = words[b * SLEN + (SLEN - 1 - t)];
  dst[((size_t)t * BSZ + b) * WDIM + cix] = (bf16_t)emb[(size_t)w * WDIM + cix];
}

__global__ void k_gather_actions(bf16_t* dst, const int* actions, const float* emb) {
  int t = blockIdx.x, b = blockIdx.y, cix = threadIdx.x;
  int a = actions[b * ALEN + t];
  dst[((size_t)t * BSZ + b) * ADIM + cix] = (bf16_t)emb[(size_t)a * ADIM + cix];
}

__global__ void k_gather_nt(bf16_t* dst, const int* actions, int t, const float* nt_emb) {
  int b = blockIdx.x, cix = threadIdx.x;
  int a = actions[b * ALEN + t] - 2;
  dst[(size_t)b * HID + cix] = (bf16_t)nt_emb[(size_t)a * HID + cix];
}

__global__ void k_bcast_guard(bf16_t* dst, const float* g) {
  int b = blockIdx.x, cix = threadIdx.x;
  dst[(size_t)b * HID + cix] = (bf16_t)g[cix];
}

__global__ void k_concat3(bf16_t* dst, const bf16_t* s0, const bf16_t* s1, const bf16_t* s2) {
  int b = blockIdx.x, cix = threadIdx.x;   // block 384
  bf16_t v;
  if (cix < 128)      v = s0[(size_t)b * 128 + cix];
  else if (cix < 256) v = s1[(size_t)b * 128 + (cix - 128)];
  else                v = s2[(size_t)b * 128 + (cix - 256)];
  dst[(size_t)b * 384 + cix] = v;
}

__global__ void k_concat2(bf16_t* dst, const bf16_t* s0, const bf16_t* s1) {
  int b = blockIdx.x, cix = threadIdx.x;   // block 256
  bf16_t v = (cix < 128) ? s0[(size_t)b * 128 + cix] : s1[(size_t)b * 128 + (cix - 128)];
  dst[(size_t)b * 256 + cix] = v;
}

// loss -= mean_b( logsoftmax(scores)[b, actions[b,t]] ). One block of 512.
__global__ void k_loss(const float* __restrict__ scores, const int* __restrict__ actions,
                       int t, float* loss) {
  __shared__ float red[BSZ];
  int b = threadIdx.x;
  const float* row = scores + (size_t)b * NACTP;
  float mx = row[0];
  for (int j = 1; j < NACT; ++j) mx = fmaxf(mx, row[j]);
  float s = 0.0f;
  for (int j = 0; j < NACT; ++j) s += __expf(row[j] - mx);
  int a = actions[b * ALEN + t];
  red[b] = (row[a] - mx) - __logf(s);
  __syncthreads();
  for (int off = 256; off > 0; off >>= 1) {
    if (b < off) red[b] += red[b + off];
    __syncthreads();
  }
  if (b == 0) loss[0] = loss[0] - red[0] / (float)BSZ;
}

// ---------------------------------------------------------------------------
// Host-side orchestration
// ---------------------------------------------------------------------------
struct Lstm2 {
  const bf16_t *Wih0, *Whh0, *Wih1, *Whh1;
  const float  *bih0, *bhh0, *bih1, *bhh1;
  int Kin;
};
struct Scratch {
  float  *c0, *c1;
  bf16_t *hA, *hB, *hC, *hD, *zeroH;
};

static const bf16_t* run_lstm2(hipStream_t s, const Scratch& S, const Lstm2& w,
                               const bf16_t* const* xs, int T,
                               bf16_t* outSeq, bf16_t* finalOut) {
  hipMemsetAsync(S.c0, 0, BH * sizeof(float), s);
  hipMemsetAsync(S.c1, 0, BH * sizeof(float), s);
  const bf16_t* h0prev = S.zeroH;
  const bf16_t* h1prev = S.zeroH;
  bf16_t* h0bufs[2] = { S.hA, S.hB };
  bf16_t* h1bufs[2] = { S.hC, S.hD };
  const bf16_t* last = S.zeroH;
  dim3 grid(8, 8);
  for (int t = 0; t < T; ++t) {
    bf16_t* h0out = h0bufs[t & 1];
    lstm_step_wmma<<<grid, 128, 0, s>>>(xs[t], w.Kin, h0prev, w.Wih0, w.Whh0,
                                        w.bih0, w.bhh0, S.c0, h0out);
    bf16_t* h1out;
    if (outSeq)                       h1out = outSeq + (size_t)t * BH;
    else if (finalOut && t == T - 1)  h1out = finalOut;
    else                              h1out = h1bufs[t & 1];
    lstm_step_wmma<<<grid, 128, 0, s>>>(h0out, HID, h1prev, w.Wih1, w.Whh1,
                                        w.bih1, w.bhh1, S.c1, h1out);
    h0prev = h0out; h1prev = h1out; last = h1out;
  }
  return last;
}

extern "C" void kernel_launch(void* const* d_in, const int* in_sizes, int n_in,
                              void* d_out, int out_size, void* d_ws, size_t ws_size,
                              hipStream_t stream) {
  (void)in_sizes; (void)n_in; (void)out_size; (void)ws_size;

  // ---- input leaf order (setup_inputs insertion order; per-dict insertion order)
  const int*   words     = (const int*)  d_in[0];
  const int*   actions   = (const int*)  d_in[1];
  const float* word_emb  = (const float*)d_in[2];
  const float* nt_emb    = (const float*)d_in[3];
  const float* act_emb   = (const float*)d_in[4];
  // 5:buf2stack.W 6:buf2stack.b 7:nt2stack.W 8:nt2stack.b
  // 9..16  subtree_fwd  (L0: Wih,Whh,bih,bhh; L1: same)
  // 17..24 subtree_bwd
  // 25..28 subtree_mlp (W1,b1,W2,b2)
  // 29..36 buffer_lstm  37..44 history_lstm  45..52 stack_lstm
  // 53..56 action_mlp (W1,b1,W2,b2)
  // 57 buffer_guard 58 history_guard 59 stack_guard
  auto F = [&](int i) { return (const float*)d_in[i]; };
  float* loss = (float*)d_out;

  // ---- deterministic bump allocator over d_ws
  char* wsp = (char*)d_ws;
  auto allocB = [&](size_t bytes) -> void* {
    void* p = wsp; wsp += (bytes + 255) & ~(size_t)255; return p;
  };
  auto aB = [&](size_t n) { return (bf16_t*)allocB(n * sizeof(bf16_t)); };
  auto aF = [&](size_t n) { return (float*) allocB(n * sizeof(float)); };

  auto conv = [&](int idx, size_t n) {
    bf16_t* d = aB(n);
    k_f32_to_bf16<<<dim3((unsigned)((n + 255) / 256)), 256, 0, stream>>>(d, F(idx), (int)n);
    return d;
  };
  auto mkLstm = [&](int base, int Kin) {
    Lstm2 w;
    w.Wih0 = conv(base + 0, (size_t)512 * Kin);
    w.Whh0 = conv(base + 1, (size_t)512 * HID);
    w.bih0 = F(base + 2); w.bhh0 = F(base + 3);
    w.Wih1 = conv(base + 4, (size_t)512 * HID);
    w.Whh1 = conv(base + 5, (size_t)512 * HID);
    w.bih1 = F(base + 6); w.bhh1 = F(base + 7);
    w.Kin = Kin;
    return w;
  };

  hipMemsetAsync(loss, 0, sizeof(float), stream);   // fresh accumulator each call

  // ---- weights -> bf16
  bf16_t* wB2S   = conv(5,  (size_t)HID * WDIM);
  bf16_t* wN2S   = conv(7,  (size_t)HID * HID);
  Lstm2 subF  = mkLstm(9,  HID);
  Lstm2 subB  = mkLstm(17, HID);
  bf16_t* wSubW1 = conv(25, (size_t)HID * 2 * HID);
  bf16_t* wSubW2 = conv(27, (size_t)HID * HID);
  Lstm2 bufL  = mkLstm(29, WDIM);
  Lstm2 hisL  = mkLstm(37, ADIM);
  Lstm2 stkL  = mkLstm(45, HID);
  bf16_t* wActW1 = conv(53, (size_t)HID * 3 * HID);
  bf16_t* wActW2 = aB((size_t)NACTP * HID);   // padded 80x128, zero rows >= 66
  {
    int n = NACTP * HID;
    k_conv_pad<<<dim3((n + 255) / 256), 256, 0, stream>>>(wActW2, F(55), NACT, NACTP, HID);
  }

  // ---- activations / scratch
  bf16_t* buff    = aB((size_t)SLEN * BSZ * WDIM);
  bf16_t* ainp    = aB((size_t)ALEN * BSZ * ADIM);
  bf16_t* buffEnc = aB((size_t)SLEN * BH);
  bf16_t* histEnc = aB((size_t)ALEN * BH);
  bf16_t* slots   = aB((size_t)16 * BH);       // stack slots (max depth 14)
  bf16_t* gBuf = aB(BH), *gHist = aB(BH), *gStk = aB(BH);
  bf16_t* ntbuf  = aB(BH);
  bf16_t* psbuf  = aB((size_t)BSZ * 3 * HID);
  bf16_t* mlph   = aB(BH);
  float*  scores = aF((size_t)BSZ * NACTP);
  bf16_t* ofbuf  = aB(BH), *obbuf = aB(BH);
  bf16_t* catbuf = aB((size_t)BSZ * 2 * HID);
  bf16_t* smid   = aB(BH);
  bf16_t* stkOut = aB(BH);
  Scratch S;
  S.c0 = aF(BH); S.c1 = aF(BH);
  S.hA = aB(BH); S.hB = aB(BH); S.hC = aB(BH); S.hD = aB(BH);
  S.zeroH = aB(BH);
  hipMemsetAsync(S.zeroH, 0, BH * sizeof(bf16_t), stream);

  // ---- embeddings / guards
  k_gather_words  <<<dim3(SLEN, BSZ), WDIM, 0, stream>>>(buff, words, word_emb);
  k_gather_actions<<<dim3(ALEN, BSZ), ADIM, 0, stream>>>(ainp, actions, act_emb);
  k_bcast_guard<<<BSZ, HID, 0, stream>>>(gBuf,  F(57));
  k_bcast_guard<<<BSZ, HID, 0, stream>>>(gHist, F(58));
  k_bcast_guard<<<BSZ, HID, 0, stream>>>(gStk,  F(59));

  // ---- buffer & history encoders
  {
    std::vector<const bf16_t*> xs(SLEN);
    for (int t = 0; t < SLEN; ++t) xs[t] = buff + (size_t)t * BSZ * WDIM;
    run_lstm2(stream, S, bufL, xs.data(), SLEN, buffEnc, nullptr);
  }
  {
    std::vector<const bf16_t*> xs(ALEN);
    for (int t = 0; t < ALEN; ++t) xs[t] = ainp + (size_t)t * BSZ * ADIM;
    run_lstm2(stream, S, hisL, xs.data(), ALEN, histEnc, nullptr);
  }

  // ---- rebuild the deterministic action sequence (REDUCE=0, SHIFT=1, NT=2)
  std::vector<int> seq; seq.push_back(2);
  for (int g = 0; g < 5; ++g) {
    seq.push_back(2);
    for (int j = 0; j < 8; ++j) seq.push_back(1);
    seq.push_back(0);
  }
  seq.push_back(0);   // ALEN = 52

  auto slotPtr = [&](int i) { return slots + (size_t)i * BH; };
  std::vector<int>  st;       // stack slot indices (bottom..top)
  std::vector<char> open;     // open-NT flags
  int buff_len = SLEN;

  for (int t = 0; t < ALEN; ++t) {
    const int a0 = seq[t];
    const bf16_t* bufState = buff_len > 0 ? buffEnc + (size_t)(buff_len - 1) * BH : gBuf;
    const bf16_t* hisState = t > 0        ? histEnc + (size_t)(t - 1) * BH        : gHist;
    const bf16_t* stkState;
    if (!st.empty()) {
      std::vector<const bf16_t*> xs(st.size());
      for (size_t i = 0; i < st.size(); ++i) xs[i] = slotPtr(st[i]);
      stkState = run_lstm2(stream, S, stkL, xs.data(), (int)st.size(), nullptr, stkOut);
    } else stkState = gStk;

    // action scores + loss
    k_concat3<<<BSZ, 384, 0, stream>>>(psbuf, bufState, hisState, stkState);
    gemm_wmma<bf16_t, true ><<<dim3(8, 8), 128, 0, stream>>>(psbuf, wActW1, F(54), mlph,
                                                             3 * HID, HID, HID);
    gemm_wmma<float,  false><<<dim3(8, 5), 128, 0, stream>>>(mlph, wActW2, F(56), scores,
                                                             HID, NACT, NACTP);
    k_loss<<<1, BSZ, 0, stream>>>(scores, actions, t, loss);

    if (a0 == 0) {              // REDUCE
      std::vector<int> kids;
      while (!open.empty() && !open.back()) { kids.push_back(st.back()); st.pop_back(); open.pop_back(); }
      int parent = st.back(); st.pop_back(); open.pop_back();
      int n = 1 + (int)kids.size();
      std::vector<const bf16_t*> xf(n), xb(n);
      xf[0] = xb[0] = slotPtr(parent);
      for (int i = 0; i < (int)kids.size(); ++i) {
        xf[1 + i] = slotPtr(kids[i]);                          // pop order
        xb[1 + i] = slotPtr(kids[kids.size() - 1 - i]);        // reversed
      }
      run_lstm2(stream, S, subF, xf.data(), n, nullptr, ofbuf);
      run_lstm2(stream, S, subB, xb.data(), n, nullptr, obbuf);
      k_concat2<<<BSZ, 256, 0, stream>>>(catbuf, ofbuf, obbuf);
      gemm_wmma<bf16_t, true ><<<dim3(8, 8), 128, 0, stream>>>(catbuf, wSubW1, F(26), smid,
                                                               2 * HID, HID, HID);
      int dst = (int)st.size();
      gemm_wmma<bf16_t, false><<<dim3(8, 8), 128, 0, stream>>>(smid, wSubW2, F(28), slotPtr(dst),
                                                               HID, HID, HID);
      st.push_back(dst); open.push_back(0);
    } else if (a0 == 1) {       // SHIFT
      int dst = (int)st.size();
      gemm_wmma<bf16_t, false><<<dim3(8, 8), 128, 0, stream>>>(
          buff + (size_t)(buff_len - 1) * BSZ * WDIM, wB2S, F(6), slotPtr(dst),
          WDIM, HID, HID);
      st.push_back(dst); open.push_back(0);
      --buff_len;
    } else {                    // NT
      k_gather_nt<<<BSZ, HID, 0, stream>>>(ntbuf, actions, t, nt_emb);
      int dst = (int)st.size();
      gemm_wmma<bf16_t, false><<<dim3(8, 8), 128, 0, stream>>>(ntbuf, wN2S, F(8), slotPtr(dst),
                                                               HID, HID, HID);
      st.push_back(dst); open.push_back(2);   // open NT
    }
  }
}
